// GATblock_352187318738
// MI455X (gfx1250) — compile-verified, matrix operand order checked by reading, not run on hardware
//
#include <hip/hip_runtime.h>
#include <hip/hip_bf16.h>
#include <stdint.h>

// ---------------------------------------------------------------------------
// TransformerConv x3 (GAT-style) for MI455X / gfx1250.
// Dense projections via V_WMMA_F32_16X16X32_BF16 (bf16 in, f32 accum),
// with the fused W^T staged in LDS (320KB/WGP) and A fragments held in
// registers across all 32 N-tiles.  Edge attention via gather +
// u32-ordered atomicMax / f32 atomicAdd.
// ---------------------------------------------------------------------------

#define NN   50000
#define EE   400000
#define HC   128          // heads * channels per layer (== F_IN)
#define QKVS 512          // fused q|k|v|skip output width

typedef __attribute__((ext_vector_type(16))) __bf16 v16bf;
typedef __attribute__((ext_vector_type(8)))  __bf16 v8bf;
typedef __attribute__((ext_vector_type(8)))  float  v8f;

__device__ __forceinline__ unsigned short f2bf(float f) {
  unsigned u = __float_as_uint(f);
  u += 0x7FFFu + ((u >> 16) & 1u);          // round-to-nearest-even
  return (unsigned short)(u >> 16);
}
__device__ __forceinline__ float bf2f(unsigned short h) {
  return __uint_as_float(((unsigned)h) << 16);
}
// order-preserving float<->uint mapping for atomicMax on floats
__device__ __forceinline__ unsigned ordf(float f) {
  unsigned u = __float_as_uint(f);
  return (u & 0x80000000u) ? ~u : (u | 0x80000000u);
}
__device__ __forceinline__ float unordf(unsigned e) {
  unsigned u = (e & 0x80000000u) ? (e & 0x7FFFFFFFu) : ~e;
  return __uint_as_float(u);
}

// ---------------------------------------------------------------- utilities
__global__ void k_fill_u32(unsigned* __restrict__ p, unsigned v, int n) {
  int i = blockIdx.x * blockDim.x + threadIdx.x;
  if (i < n) p[i] = v;
}
__global__ void k_f32_to_bf16(const float* __restrict__ in,
                              unsigned short* __restrict__ out, int n) {
  int i = blockIdx.x * blockDim.x + threadIdx.x;
  if (i < n) out[i] = f2bf(in[i]);
}
// W [din x dout] fp32 (row-major)  ->  Wt [dout x din] bf16
__global__ void k_transpose_bf16(const float* __restrict__ in,
                                 unsigned short* __restrict__ out,
                                 int din, int dout) {
  int i = blockIdx.x * blockDim.x + threadIdx.x;
  if (i < din * dout) {
    int r = i / dout, c = i % dout;
    out[(size_t)c * din + r] = f2bf(in[i]);
  }
}
__global__ void k_copy_f32(const float* __restrict__ in, float* __restrict__ out, int n) {
  int i = blockIdx.x * blockDim.x + threadIdx.x;
  if (i < n) out[i] = in[i];
}

// ------------------------------------------------- fused node GEMM via WMMA
// Y[N x 512] = Xb[N x 128] @ WtFused^T + bias.
// Block = 8 waves.  Block stages entire fused W^T (512x128 bf16 = 128 KB)
// into LDS; each wave owns one 16-row M-tile, keeps its A fragments (K=128,
// 4 x v16bf) in VGPRs, and sweeps all 32 N-tiles reading B from LDS.
// A-fragment layout per lane (16-bit 16x32): row m0+(L&15);
// K chunks {k0+off..+8, k0+16+off..+8}, off = 8*(L>=16).
__global__ void __launch_bounds__(256) node_gemm_wmma(
    const unsigned short* __restrict__ Xb,
    const unsigned short* __restrict__ Wt,   // [512 x 128] bf16 (out-ch major)
    const float* __restrict__ bias,          // [512]
    float* __restrict__ Y,                   // [N x 512]
    int nrows) {
  extern __shared__ unsigned short ldsW[];   // QKVS*HC bf16 = 128 KB
  // cooperative stage: 8192 x 16B chunks over 256 threads
  {
    const uint4* sp = reinterpret_cast<const uint4*>(Wt);
    uint4*       dp = reinterpret_cast<uint4*>(ldsW);
#pragma unroll
    for (int i = 0; i < (QKVS * HC) / (8 * 256); ++i)
      dp[i * 256 + threadIdx.x] = sp[i * 256 + threadIdx.x];
  }
  __syncthreads();

  const int lane = threadIdx.x & 31;
  const int wv   = threadIdx.x >> 5;
  const int mt   = blockIdx.x * 8 + wv;
  if (mt * 16 < nrows) {                     // wave-uniform: EXEC all-ones at WMMA
    const int row  = mt * 16 + (lane & 15);
    const int koff = (lane >> 4) * 8;
    const unsigned short* ap = Xb + (size_t)row * 128 + koff;
    union U { v16bf v; v8bf h[2]; };
    U A[4];
#pragma unroll
    for (int kk = 0; kk < 4; ++kk) {
      A[kk].h[0] = *reinterpret_cast<const v8bf*>(ap + kk * 32);
      A[kk].h[1] = *reinterpret_cast<const v8bf*>(ap + kk * 32 + 16);
    }
    const int colr = lane & 15;
    // C/D layout: VGPR j -> M = j + 8*(L>=16), N = lane%16
    float* ybase = Y + (size_t)(mt * 16 + ((lane >> 4) * 8)) * QKVS + colr;
    const unsigned short* bbase = ldsW + colr * 128 + koff;
#pragma unroll 2
    for (int tn = 0; tn < 32; ++tn) {
      const unsigned short* bp = bbase + tn * 16 * 128;
      v8f acc = {};
#pragma unroll
      for (int kk = 0; kk < 4; ++kk) {
        U B;
        B.h[0] = *reinterpret_cast<const v8bf*>(bp + kk * 32);       // ds_load_b128
        B.h[1] = *reinterpret_cast<const v8bf*>(bp + kk * 32 + 16);
        acc = __builtin_amdgcn_wmma_f32_16x16x32_bf16(false, A[kk].v, false, B.v,
                                                      (short)0, acc, false, false);
      }
      const float bn = bias[tn * 16 + colr];
      float* yp = ybase + tn * 16;
#pragma unroll
      for (int j = 0; j < 8; ++j)
        yp[(size_t)j * QKVS] = acc[j] + bn;  // constant imm offsets (dout=512)
    }
  }
}

// ----------------------------------------------- edge projection via WMMA
// e[E x 128] (bf16) = ea[E x 16] @ We[16 x 128]; K padded 16->32 with zeros.
__global__ void __launch_bounds__(256) edge_gemm_wmma(
    const unsigned short* __restrict__ Eb,
    const unsigned short* __restrict__ WeT, // [128 x 16]
    unsigned short* __restrict__ Yb, int nrows) {
  const int lane = threadIdx.x & 31;
  const int wave = blockIdx.x * (blockDim.x >> 5) + (threadIdx.x >> 5);
  const int tiles_n = HC >> 4;                     // 8
  const int tiles = (nrows >> 4) * tiles_n;
  if (wave >= tiles) return;
  const int tm = wave / tiles_n, tn = wave % tiles_n;
  const int row  = tm * 16 + (lane & 15);
  const int col  = tn * 16 + (lane & 15);
  const int koff = (lane >> 4) * 8;
  union U { v16bf v; v8bf h[2]; unsigned short u[16]; };
  U A, B;
  A.h[0] = *reinterpret_cast<const v8bf*>(Eb  + (size_t)row * 16 + koff);
  B.h[0] = *reinterpret_cast<const v8bf*>(WeT + (size_t)col * 16 + koff);
#pragma unroll
  for (int j = 0; j < 8; ++j) { A.u[8 + j] = 0; B.u[8 + j] = 0; }  // K 16..31 zero
  v8f acc = {};
  acc = __builtin_amdgcn_wmma_f32_16x16x32_bf16(false, A.v, false, B.v,
                                                (short)0, acc, false, false);
  const int rbase = tm * 16 + ((lane >> 4) * 8);
#pragma unroll
  for (int j = 0; j < 8; ++j)
    Yb[(size_t)(rbase + j) * HC + col] = f2bf(acc[j]);
}

// ------------------------------------------------------- attention (edges)
__global__ void k_alpha(const float* __restrict__ qkvs,
                        const unsigned short* __restrict__ eproj,
                        const long long* __restrict__ src,
                        const long long* __restrict__ dst,
                        float* __restrict__ alpha, unsigned* __restrict__ menc,
                        int ne) {
  int e = blockIdx.x * blockDim.x + threadIdx.x;
  if (e >= ne) return;
  const int s = (int)src[e], d = (int)dst[e];
  const float* qd = qkvs + (size_t)d * QKVS;          // q at +0
  const float* ks = qkvs + (size_t)s * QKVS + 128;    // k at +128
  const unsigned short* ep = eproj + (size_t)e * HC;
  float a0 = 0.f, a1 = 0.f, a2 = 0.f, a3 = 0.f;
#pragma unroll
  for (int c = 0; c < 32; ++c) {
    a0 += qd[c]      * (ks[c]      + bf2f(ep[c]));
    a1 += qd[32 + c] * (ks[32 + c] + bf2f(ep[32 + c]));
    a2 += qd[64 + c] * (ks[64 + c] + bf2f(ep[64 + c]));
    a3 += qd[96 + c] * (ks[96 + c] + bf2f(ep[96 + c]));
  }
  const float sc = 0.17677669529663687f;              // 1/sqrt(32)
  a0 *= sc; a1 *= sc; a2 *= sc; a3 *= sc;
  float* ap = alpha + (size_t)e * 4;
  ap[0] = a0; ap[1] = a1; ap[2] = a2; ap[3] = a3;
  unsigned* mp = menc + (size_t)d * 4;
  atomicMax(mp + 0, ordf(a0));
  atomicMax(mp + 1, ordf(a1));
  atomicMax(mp + 2, ordf(a2));
  atomicMax(mp + 3, ordf(a3));
}

__global__ void k_expsum(const long long* __restrict__ dst,
                         const unsigned* __restrict__ menc,
                         float* __restrict__ alpha, float* __restrict__ ssum,
                         int n4) {
  int t = blockIdx.x * blockDim.x + threadIdx.x;
  if (t >= n4) return;
  const int e = t >> 2, h = t & 3;
  const int d = (int)dst[e];
  const unsigned enc = menc[(size_t)d * 4 + h];
  const float m = enc ? unordf(enc) : 0.0f;           // matches where(isfinite(m), m, 0)
  const float ex = __expf(alpha[t] - m);
  alpha[t] = ex;
  atomicAdd(&ssum[(size_t)d * 4 + h], ex);
}

__global__ void k_scatter(const float* __restrict__ qkvs,
                          const unsigned short* __restrict__ eproj,
                          const long long* __restrict__ src,
                          const long long* __restrict__ dst,
                          const float* __restrict__ alpha,
                          const float* __restrict__ ssum,
                          float* __restrict__ agg, long total) {
  long t = (long)blockIdx.x * blockDim.x + threadIdx.x;
  if (t >= total) return;
  const int e = (int)(t >> 7), ch = (int)(t & 127), h = ch >> 5;
  const int d = (int)dst[e], s = (int)src[e];
  const float w  = alpha[(size_t)e * 4 + h] / (ssum[(size_t)d * 4 + h] + 1e-16f);
  const float vj = qkvs[(size_t)s * QKVS + 256 + ch] + bf2f(eproj[(size_t)e * HC + ch]);
  atomicAdd(&agg[(size_t)d * HC + ch], w * vj);
}

__global__ void k_addskip(const float* __restrict__ qkvs, float* __restrict__ agg, int n) {
  int t = blockIdx.x * blockDim.x + threadIdx.x;
  if (t >= n) return;
  const int node = t >> 7, ch = t & 127;
  agg[t] += qkvs[(size_t)node * QKVS + 384 + ch];     // skip at +384
}

// ------------------------------------------------------------ batchnorm
__global__ void k_bnstats(const float* __restrict__ xin, float* __restrict__ sums,
                          int nrows) {
  const int ch = threadIdx.x;                         // blockDim == 128
  const int r0 = blockIdx.x * 64;
  float s = 0.f, s2 = 0.f;
  for (int r = 0; r < 64; ++r) {
    const int row = r0 + r;
    if (row < nrows) {
      const float v = xin[(size_t)row * HC + ch];
      s += v; s2 += v * v;
    }
  }
  atomicAdd(&sums[ch], s);
  atomicAdd(&sums[HC + ch], s2);
}

__global__ void k_bnapply(const float* __restrict__ xin, const float* __restrict__ sums,
                          const float* __restrict__ gamma, const float* __restrict__ beta,
                          float* __restrict__ yout, unsigned short* __restrict__ ybf,
                          int n) {
  int t = blockIdx.x * blockDim.x + threadIdx.x;
  if (t >= n) return;
  const int ch = t & 127;
  const float invN = 1.0f / (float)NN;
  const float mu  = sums[ch] * invN;
  const float var = sums[HC + ch] * invN - mu * mu;   // biased var, like jnp.var
  float y = (xin[t] - mu) * rsqrtf(var + 1e-5f) * gamma[ch] + beta[ch];
  y = fmaxf(y, 0.0f);
  yout[t] = y;
  ybf[t]  = f2bf(y);                                  // next layer's WMMA input
}

// ---------------------------------------------------------------- driver
extern "C" void kernel_launch(void* const* d_in, const int* in_sizes, int n_in,
                              void* d_out, int out_size, void* d_ws, size_t ws_size,
                              hipStream_t stream) {
  (void)in_sizes; (void)n_in; (void)out_size;
  // jax pytree flatten order (dicts sorted by key):
  //  0:x  1:edge_index(i64)  2:edge_weigth
  //  3/4: bn1 beta/gamma  5/6: bn2  7/8: bn3
  //  9..17:  l1 {We,Wk,Wq,Ws,Wv,bk,bq,bs,bv}; 18..26: l2; 27..35: l3
  const float*     x  = (const float*)d_in[0];
  const long long* ei = (const long long*)d_in[1];
  const float*     ea = (const float*)d_in[2];
  const float* bn_beta[3]  = {(const float*)d_in[3], (const float*)d_in[5], (const float*)d_in[7]};
  const float* bn_gamma[3] = {(const float*)d_in[4], (const float*)d_in[6], (const float*)d_in[8]};
  struct LP { const float *We, *Wk, *Wq, *Ws, *Wv, *bk, *bq, *bs, *bv; };
  LP lp[3];
  for (int l = 0; l < 3; ++l) {
    const int b = 9 + l * 9;
    lp[l].We = (const float*)d_in[b + 0]; lp[l].Wk = (const float*)d_in[b + 1];
    lp[l].Wq = (const float*)d_in[b + 2]; lp[l].Ws = (const float*)d_in[b + 3];
    lp[l].Wv = (const float*)d_in[b + 4]; lp[l].bk = (const float*)d_in[b + 5];
    lp[l].bq = (const float*)d_in[b + 6]; lp[l].bs = (const float*)d_in[b + 7];
    lp[l].bv = (const float*)d_in[b + 8];
  }

  uint8_t* ws = (uint8_t*)d_ws;
  size_t off = 0;
  auto alloc = [&](size_t bytes) -> void* {
    void* p = ws + off;
    off = (off + bytes + 255) & ~(size_t)255;
    return p;
  };
  unsigned short* xb    = (unsigned short*)alloc((size_t)NN * HC * 2);   // x (bf16), reused between layers
  unsigned short* eab   = (unsigned short*)alloc((size_t)EE * 16 * 2);   // edge attrs bf16
  unsigned short* wt    = (unsigned short*)alloc((size_t)3 * QKVS * HC * 2); // fused W^T bf16
  unsigned short* wet   = (unsigned short*)alloc((size_t)3 * HC * 16 * 2);   // We^T bf16
  float*          biasf = (float*)alloc((size_t)3 * QKVS * 4);
  float*          qkvs  = (float*)alloc((size_t)NN * QKVS * 4);          // q|k|v|skip
  unsigned short* eproj = (unsigned short*)alloc((size_t)EE * HC * 2);   // edge proj bf16
  float*          alpha = (float*)alloc((size_t)EE * 4 * 4);
  unsigned*       menc  = (unsigned*)alloc((size_t)NN * 4 * 4);
  float*          ssum  = (float*)alloc((size_t)NN * 4 * 4);
  float*          agg   = (float*)alloc((size_t)NN * HC * 4);
  float*          bnsum = (float*)alloc((size_t)2 * HC * 4);
  if (off > ws_size) return;   // scratch too small: do nothing (deterministic)

  const long long* srcp = ei;
  const long long* dstp = ei + EE;

  k_f32_to_bf16<<<(NN * HC + 255) / 256, 256, 0, stream>>>(x, xb, NN * HC);
  k_f32_to_bf16<<<(EE * 16 + 255) / 256, 256, 0, stream>>>(ea, eab, EE * 16);

  // weight prep: transpose to [out x K] bf16, fused q|k|v|s per layer
  for (int l = 0; l < 3; ++l) {
    unsigned short* wtl  = wt  + (size_t)l * QKVS * HC;
    unsigned short* wetl = wet + (size_t)l * HC * 16;
    float*          bfl  = biasf + (size_t)l * QKVS;
    const float* Wm[4] = {lp[l].Wq, lp[l].Wk, lp[l].Wv, lp[l].Ws};
    const float* bm[4] = {lp[l].bq, lp[l].bk, lp[l].bv, lp[l].bs};
    for (int m = 0; m < 4; ++m) {
      k_transpose_bf16<<<(HC * HC + 255) / 256, 256, 0, stream>>>(
          Wm[m], wtl + (size_t)m * HC * HC, HC, HC);
      k_copy_f32<<<1, 128, 0, stream>>>(bm[m], bfl + m * HC, HC);
    }
    k_transpose_bf16<<<(16 * HC + 255) / 256, 256, 0, stream>>>(lp[l].We, wetl, 16, HC);
  }

  const int node_blocks = (NN / 16 + 7) / 8;        // 8 M-tiles per block
  const int edge_waves  = (EE / 16) * (HC / 16);    // 200000
  const size_t lds_bytes = (size_t)QKVS * HC * 2;   // 128 KB dynamic LDS

  for (int l = 0; l < 3; ++l) {
    unsigned short* wtl  = wt  + (size_t)l * QKVS * HC;
    unsigned short* wetl = wet + (size_t)l * HC * 16;
    float*          bfl  = biasf + (size_t)l * QKVS;

    node_gemm_wmma<<<node_blocks, 256, lds_bytes, stream>>>(xb, wtl, bfl, qkvs, NN);
    edge_gemm_wmma<<<edge_waves / 8, 256, 0, stream>>>(eab, wetl, eproj, EE);

    k_fill_u32<<<(NN * 4 + 255) / 256, 256, 0, stream>>>(menc, 0u, NN * 4);
    k_fill_u32<<<(NN * 4 + 255) / 256, 256, 0, stream>>>((unsigned*)ssum, 0u, NN * 4);
    k_fill_u32<<<(NN * HC + 255) / 256, 256, 0, stream>>>((unsigned*)agg, 0u, NN * HC);

    k_alpha<<<(EE + 255) / 256, 256, 0, stream>>>(qkvs, eproj, srcp, dstp, alpha, menc, EE);
    k_expsum<<<(EE * 4 + 255) / 256, 256, 0, stream>>>(dstp, menc, alpha, ssum, EE * 4);

    const long tot = (long)EE * HC;
    k_scatter<<<(unsigned)((tot + 255) / 256), 256, 0, stream>>>(
        qkvs, eproj, srcp, dstp, alpha, ssum, agg, tot);

    k_addskip<<<(NN * HC + 255) / 256, 256, 0, stream>>>(qkvs, agg, NN * HC);

    k_fill_u32<<<1, 256, 0, stream>>>((unsigned*)bnsum, 0u, 2 * HC);
    k_bnstats<<<(NN + 63) / 64, 128, 0, stream>>>(agg, bnsum, NN);

    float* yout = (float*)d_out + (size_t)l * NN * HC;
    k_bnapply<<<(NN * HC + 255) / 256, 256, 0, stream>>>(
        agg, bnsum, bn_gamma[l], bn_beta[l], yout, xb, NN * HC);
  }
}
